// ToyACmem_2723009266593
// MI455X (gfx1250) — compile-verified, matrix operand order checked by reading, not run on hardware
//
#include <hip/hip_runtime.h>

// ---------------- problem sizes ----------------
#define BSN 2048   // batch (sequential scan)
#define MM  256    // memory slots
#define EE  128    // memory emb
#define HSN 512    // summarizer hidden
#define HAN 512    // AC hidden
#define HPN 256    // processor hidden
#define TROWS 272  // token rows 258 padded to 17*16
#define KP    288  // G5 K dim 258 padded to 9*32

typedef __attribute__((ext_vector_type(8)))  __bf16 v8bf;
typedef __attribute__((ext_vector_type(16))) __bf16 v16bf;
typedef __attribute__((ext_vector_type(8)))  float  v8f;

// ---- bf16 <-> f32 as raw ushort bits (RNE) ----
__device__ __forceinline__ unsigned short f2u(float x) {
    unsigned u = __float_as_uint(x);
    unsigned r = (u + 0x7FFFu + ((u >> 16) & 1u)) >> 16;
    return (unsigned short)r;
}
__device__ __forceinline__ float u2f(unsigned short h) {
    return __uint_as_float(((unsigned)h) << 16);
}

__device__ __forceinline__ v8f wmma_bf16(v16bf a, v16bf b, v8f c) {
    return __builtin_amdgcn_wmma_f32_16x16x32_bf16(false, a, false, b, (short)0, c, false, false);
}

// A-frag (16x32 bf16) from row-major bf16 buffer (LDS or global).
// lane<16: M=lane,  K in {0..7,16..23}; lane>=16: M=lane-16, K in {8..15,24..31}
__device__ __forceinline__ v16bf ld_afrag(const unsigned short* base, int rowBase,
                                          int ld, int kOff, int lane) {
    int r  = rowBase + (lane & 15);
    int kb = kOff + ((lane < 16) ? 0 : 8);
    v8bf lo = *reinterpret_cast<const v8bf*>(base + r * ld + kb);
    v8bf hi = *reinterpret_cast<const v8bf*>(base + r * ld + kb + 16);
    return __builtin_shufflevector(lo, hi, 0,1,2,3,4,5,6,7,8,9,10,11,12,13,14,15);
}

// B-frag (32x16 bf16) from pre-packed global weights: [(kt*NT+nt)*32 + lane]*16 contiguous
__device__ __forceinline__ v16bf ld_bfrag_packed(const unsigned short* Wp, int kt, int nt,
                                                 int NT, int lane) {
    return *reinterpret_cast<const v16bf*>(Wp + ((size_t)(kt * NT + nt) * 32 + lane) * 16);
}

// B-frag (32x16 bf16) gathered from a row-major [rows x ld] bf16 buffer (K = row index)
__device__ __forceinline__ v16bf ld_bfrag_rm(const unsigned short* base, int kt, int nBase,
                                             int ld, int rowMax, int lane) {
    int n  = nBase + (lane & 15);
    int kb = (lane < 16) ? 0 : 8;
    v16bf b;
#pragma unroll
    for (int i = 0; i < 16; ++i) {
        int k = kt * 32 + kb + i + ((i < 8) ? 0 : 8);
        unsigned short u = (k < rowMax) ? base[k * ld + n] : (unsigned short)0;
        __bf16 e; __builtin_memcpy(&e, &u, 2);
        b[i] = e;
    }
    return b;
}

// ---------------- setup kernels ----------------
__global__ void k_pe(float* __restrict__ pe) {
    int idx = blockIdx.x * blockDim.x + threadIdx.x;
    if (idx >= MM * EE) return;
    int pos = idx >> 7, c = idx & 127;
    int k2 = (c >> 1) << 1;
    float div = __expf(-(float)k2 * (logf(10000.f) / (float)EE));
    float arg = (float)pos * div;
    pe[idx] = (c & 1) ? cosf(arg) : sinf(arg);
}

__global__ void k_emb(const float* __restrict__ st, const float* __restrict__ W,
                      const float* __restrict__ b, float* __restrict__ se) {
    int idx = blockIdx.x * blockDim.x + threadIdx.x;
    if (idx >= BSN * EE) return;
    int bi = idx >> 7, e = idx & 127;
    float acc = b[e];
#pragma unroll
    for (int i = 0; i < 4; ++i) acc += st[bi * 4 + i] * W[i * EE + e];
    se[idx] = acc;
}

// pack row-major f32 W[K][N] into WMMA B-frag-major bf16
__global__ void k_pack(const float* __restrict__ W, int K, int N, unsigned short* __restrict__ out) {
    int idx = blockIdx.x * blockDim.x + threadIdx.x;
    int NT = N >> 4;
    int total = (K >> 5) * NT * 512;
    if (idx >= total) return;
    int i    = idx & 15;
    int lane = (idx >> 4) & 31;
    int nt   = (idx >> 9) % NT;
    int kt   = (idx >> 9) / NT;
    int k = (kt << 5) + i + ((i < 8) ? 0 : 8) + ((lane < 16) ? 0 : 8);
    int n = (nt << 4) + (lane & 15);
    out[idx] = f2u(W[k * N + n]);
}

// pack vector v[K] replicated across all 16 columns into B-frag-major bf16 (NT=1)
// D = A x B with B[k,n]=v[k] gives per-row weighted sums in every column of D.
__global__ void k_packrep(const float* __restrict__ v, int K, unsigned short* __restrict__ out) {
    int idx = blockIdx.x * blockDim.x + threadIdx.x;
    int total = (K >> 5) * 512;
    if (idx >= total) return;
    int i    = idx & 15;
    int lane = (idx >> 4) & 31;
    int kt   = idx >> 9;
    int k = (kt << 5) + i + ((i < 8) ? 0 : 8) + ((lane < 16) ? 0 : 8);
    out[idx] = f2u(v[k]);
}

// ---------------- persistent main kernel: one WGP, 16 waves ----------------
__global__ __launch_bounds__(512, 1)
void k_main(const float* __restrict__ memory0,
            const float* __restrict__ peg,
            const float* __restrict__ seg,
            const unsigned short* __restrict__ Wr1p,
            const unsigned short* __restrict__ Ww1p,
            const unsigned short* __restrict__ Ww2p,
            const unsigned short* __restrict__ Wr2p,
            unsigned short* __restrict__ wTg,
            const float* __restrict__ br1,
            const float* __restrict__ bw1, const float* __restrict__ bw2,
            const float* __restrict__ Ws,  const float* __restrict__ bs_,
            const float* __restrict__ Wa,  const float* __restrict__ ba,
            const float* __restrict__ Wc,  const float* __restrict__ bc,
            const float* __restrict__ Wp1, const float* __restrict__ bp1,
            const float* __restrict__ Wp2, const float* __restrict__ bp2,
            float* __restrict__ out)
{
    __shared__ __align__(16) unsigned short Abuf[TROWS * EE];   // [s; ms2; mem+PE; 0pad] bf16
    __shared__ __align__(16) unsigned short HWbuf[TROWS * EE];  // hidden chunk bf16 (reader & writer)
    __shared__ float wr_l[TROWS];  // reader logits / weights
    __shared__ float cs[264];      // [s(128); mem_state(128); probs(5); val(1)]
    __shared__ float hbuf[HAN];
    __shared__ float hp[HPN];
    __shared__ float red[16];
    __shared__ float sc2[8];

    const int tid  = threadIdx.x;
    const int w    = tid >> 5;
    const int lane = tid & 31;
    const int hlf  = (lane < 16) ? 0 : 8;
    const int l15  = lane & 15;

    // init: A rows 2..257 = bf16(memory0 + PE), others 0; zero wT (incl. K padding)
    for (int idx = tid; idx < TROWS * EE; idx += 512) {
        int r = idx >> 7, e = idx & 127;
        float v = 0.f;
        if (r >= 2 && r < 258) { int j = r - 2; v = memory0[j * EE + e] + peg[j * EE + e]; }
        Abuf[idx] = f2u(v);
    }
    for (int idx = tid; idx < MM * KP; idx += 512) wTg[idx] = 0;
    __threadfence();
    __syncthreads();

    for (int step = 0; step < BSN; ++step) {
        // ---- stage state row ----
        if (tid < EE) {
            float sv = seg[step * EE + tid];
            cs[tid] = sv;
            Abuf[tid] = f2u(sv);               // token row 0 = s
        }
        __syncthreads();

        // ---- G1 reader: hidden chunks + WMMA row-sum against replicated Wr2 ----
        // logit_t = sum_c relu(h[t,c]+br1[c]) * Wr2[c]; no shuffles, no atomics.
        {
            v8f accL0 = (v8f){0.f,0.f,0.f,0.f,0.f,0.f,0.f,0.f};   // mt = w
            v8f accL1 = (v8f){0.f,0.f,0.f,0.f,0.f,0.f,0.f,0.f};   // mt = 16 (wave 0 only)
#pragma unroll 1
            for (int kc = 0; kc < 4; ++kc) {
                // hidden chunk -> HWbuf
#pragma unroll 1
                for (int tile = w; tile < 136; tile += 16) {
                    int mt = tile >> 3, nl = tile & 7;
                    int ntg = (kc << 3) + nl;
                    v16bf bb0 = ld_bfrag_packed(Wr1p, 0, ntg, 32, lane);
                    v16bf bb1 = ld_bfrag_packed(Wr1p, 1, ntg, 32, lane);
                    v16bf bb2 = ld_bfrag_packed(Wr1p, 2, ntg, 32, lane);
                    v16bf bb3 = ld_bfrag_packed(Wr1p, 3, ntg, 32, lane);
                    v8f c = {0.f,0.f,0.f,0.f,0.f,0.f,0.f,0.f};
                    c = wmma_bf16(ld_afrag(Abuf, mt << 4, EE, 0,  lane), bb0, c);
                    c = wmma_bf16(ld_afrag(Abuf, mt << 4, EE, 32, lane), bb1, c);
                    c = wmma_bf16(ld_afrag(Abuf, mt << 4, EE, 64, lane), bb2, c);
                    c = wmma_bf16(ld_afrag(Abuf, mt << 4, EE, 96, lane), bb3, c);
                    int colh = (nl << 4) + l15;
                    float b1 = br1[(kc << 7) + colh];
#pragma unroll
                    for (int v = 0; v < 8; ++v) {
                        int row = (mt << 4) + hlf + v;
                        HWbuf[row * EE + colh] = f2u(fmaxf(c[v] + b1, 0.f));
                    }
                }
                __syncthreads();
                // row-sum WMMAs: B = Wr2 replicated across 16 cols
                v16bf r0 = ld_bfrag_packed(Wr2p, (kc << 2) + 0, 0, 1, lane);
                v16bf r1 = ld_bfrag_packed(Wr2p, (kc << 2) + 1, 0, 1, lane);
                v16bf r2 = ld_bfrag_packed(Wr2p, (kc << 2) + 2, 0, 1, lane);
                v16bf r3 = ld_bfrag_packed(Wr2p, (kc << 2) + 3, 0, 1, lane);
                accL0 = wmma_bf16(ld_afrag(HWbuf, w << 4, EE, 0,  lane), r0, accL0);
                accL0 = wmma_bf16(ld_afrag(HWbuf, w << 4, EE, 32, lane), r1, accL0);
                accL0 = wmma_bf16(ld_afrag(HWbuf, w << 4, EE, 64, lane), r2, accL0);
                accL0 = wmma_bf16(ld_afrag(HWbuf, w << 4, EE, 96, lane), r3, accL0);
                if (w == 0) {
                    accL1 = wmma_bf16(ld_afrag(HWbuf, 256, EE, 0,  lane), r0, accL1);
                    accL1 = wmma_bf16(ld_afrag(HWbuf, 256, EE, 32, lane), r1, accL1);
                    accL1 = wmma_bf16(ld_afrag(HWbuf, 256, EE, 64, lane), r2, accL1);
                    accL1 = wmma_bf16(ld_afrag(HWbuf, 256, EE, 96, lane), r3, accL1);
                }
                __syncthreads();   // before next chunk overwrites HWbuf
            }
            // every column of D equals the row logit; lanes with l15==0 publish
            if (l15 == 0) {
#pragma unroll
                for (int v = 0; v < 8; ++v) wr_l[(w << 4) + hlf + v] = accL0[v];
                if (w == 0) {
#pragma unroll
                    for (int v = 0; v < 8; ++v) wr_l[256 + hlf + v] = accL1[v];
                }
            }
        }
        __syncthreads();

        // ---- reader softmax over tokens {0} U [2,258)  (br2 is softmax-invariant) ----
        {
            float lv = -3.0e38f;
            bool vd = (tid == 0) || (tid >= 2 && tid < 258);
            if (tid < TROWS && vd) lv = wr_l[tid];
#pragma unroll
            for (int o = 16; o > 0; o >>= 1) lv = fmaxf(lv, __shfl_xor(lv, o));
            if (lane == 0) red[w] = lv;
            __syncthreads();
            if (tid == 0) {
                float m = red[0];
                for (int i = 1; i < 16; ++i) m = fmaxf(m, red[i]);
                sc2[6] = m;
            }
            __syncthreads();
            float mx = sc2[6];
            float ev = 0.f;
            if (tid < TROWS && vd) ev = __expf(wr_l[tid] - mx);
            if (tid < TROWS) wr_l[tid] = ev;
            float sv = ev;
#pragma unroll
            for (int o = 16; o > 0; o >>= 1) sv += __shfl_xor(sv, o);
            if (lane == 0) red[w] = sv;
            __syncthreads();
            if (tid == 0) {
                float s = 0.f;
                for (int i = 0; i < 16; ++i) s += red[i];
                sc2[7] = 1.f / s;
            }
            __syncthreads();
        }

        // ---- mem_state = sum_t w_t * x_t ----
        if (tid < EE) {
            float acc = 0.f;
            for (int t = 0; t < 258; ++t) acc += wr_l[t] * u2f(Abuf[t * EE + tid]);
            cs[EE + tid] = acc * sc2[7];
        }
        __syncthreads();

        // ---- shared trunk h = relu(cstate . Ws + bs) ----
        {
            float acc = bs_[tid];
            for (int k = 0; k < 2 * EE; ++k) acc += cs[k] * Ws[k * HAN + tid];
            hbuf[tid] = fmaxf(acc, 0.f);
        }
        __syncthreads();

        // ---- actor / critic heads (wave-per-output dots) ----
        if (w < 5) {
            float acc = 0.f;
            for (int k = lane; k < HAN; k += 32) acc += hbuf[k] * Wa[k * 5 + w];
#pragma unroll
            for (int o = 16; o > 0; o >>= 1) acc += __shfl_xor(acc, o);
            if (lane == 0) sc2[w] = acc + ba[w];
        } else if (w == 5) {
            float acc = 0.f;
            for (int k = lane; k < HAN; k += 32) acc += hbuf[k] * Wc[k];
#pragma unroll
            for (int o = 16; o > 0; o >>= 1) acc += __shfl_xor(acc, o);
            if (lane == 0) sc2[5] = acc + bc[0];
        }
        __syncthreads();
        if (tid == 0) {
            float mx = sc2[0];
            for (int i = 1; i < 5; ++i) mx = fmaxf(mx, sc2[i]);
            float p[5]; float s = 0.f;
            for (int i = 0; i < 5; ++i) { p[i] = __expf(sc2[i] - mx); s += p[i]; }
            float inv = 1.f / s;
            for (int i = 0; i < 5; ++i) { float pv = p[i] * inv; cs[2 * EE + i] = pv; out[step * 5 + i] = pv; }
            cs[2 * EE + 5] = sc2[5];
            out[BSN * 5 + step] = sc2[5];
        }
        __syncthreads();

        // ---- processor: ms2 = relu(pin.Wp1+bp1).Wp2+bp2 -> token row 1 ----
        if (tid < HPN) {
            float acc = bp1[tid];
            for (int k = 0; k < 262; ++k) acc += cs[k] * Wp1[k * HPN + tid];
            hp[tid] = fmaxf(acc, 0.f);
        }
        __syncthreads();
        if (tid < EE) {
            float acc = bp2[tid];
            for (int k = 0; k < HPN; ++k) acc += hp[k] * Wp2[k * EE + tid];
            Abuf[EE + tid] = f2u(acc);
        }
        __syncthreads();

        // ---- writer: G3 (hidden chunks) + G4 (logits, register-resident per wave) ----
        v8f acc2[17];
#pragma unroll
        for (int m = 0; m < 17; ++m) acc2[m] = (v8f){0.f,0.f,0.f,0.f,0.f,0.f,0.f,0.f};

#pragma unroll 1
        for (int kc = 0; kc < 4; ++kc) {
            // G3: 128-column hidden chunk into HWbuf
#pragma unroll 1
            for (int tile = w; tile < 136; tile += 16) {
                int mt = tile >> 3, nl = tile & 7;
                int ntg = (kc << 3) + nl;
                v16bf bb0 = ld_bfrag_packed(Ww1p, 0, ntg, 32, lane);
                v16bf bb1 = ld_bfrag_packed(Ww1p, 1, ntg, 32, lane);
                v16bf bb2 = ld_bfrag_packed(Ww1p, 2, ntg, 32, lane);
                v16bf bb3 = ld_bfrag_packed(Ww1p, 3, ntg, 32, lane);
                v8f c = {0.f,0.f,0.f,0.f,0.f,0.f,0.f,0.f};
                c = wmma_bf16(ld_afrag(Abuf, mt << 4, EE, 0,  lane), bb0, c);
                c = wmma_bf16(ld_afrag(Abuf, mt << 4, EE, 32, lane), bb1, c);
                c = wmma_bf16(ld_afrag(Abuf, mt << 4, EE, 64, lane), bb2, c);
                c = wmma_bf16(ld_afrag(Abuf, mt << 4, EE, 96, lane), bb3, c);
                int colh = (nl << 4) + l15;
                float b1 = bw1[(kc << 7) + colh];
#pragma unroll
                for (int v = 0; v < 8; ++v) {
                    int row = (mt << 4) + hlf + v;
                    HWbuf[row * EE + colh] = f2u(fmaxf(c[v] + b1, 0.f));
                }
            }
            __syncthreads();
            // G4 partial: wave w owns column tile n=w, all 17 row tiles
            v16bf bq0 = ld_bfrag_packed(Ww2p, (kc << 2) + 0, w, 16, lane);
            v16bf bq1 = ld_bfrag_packed(Ww2p, (kc << 2) + 1, w, 16, lane);
            v16bf bq2 = ld_bfrag_packed(Ww2p, (kc << 2) + 2, w, 16, lane);
            v16bf bq3 = ld_bfrag_packed(Ww2p, (kc << 2) + 3, w, 16, lane);
#pragma unroll
            for (int mt = 0; mt < 17; ++mt) {
                acc2[mt] = wmma_bf16(ld_afrag(HWbuf, mt << 4, EE, 0,  lane), bq0, acc2[mt]);
                acc2[mt] = wmma_bf16(ld_afrag(HWbuf, mt << 4, EE, 32, lane), bq1, acc2[mt]);
                acc2[mt] = wmma_bf16(ld_afrag(HWbuf, mt << 4, EE, 64, lane), bq2, acc2[mt]);
                acc2[mt] = wmma_bf16(ld_afrag(HWbuf, mt << 4, EE, 96, lane), bq3, acc2[mt]);
            }
            __syncthreads();
        }

        // ---- writer column softmax (registers + shfl_xor(16)) -> transposed wT bf16 ----
        {
            int col = (w << 4) + l15;
            float bcl = bw2[col];
            float mx = -3.0e38f;
#pragma unroll
            for (int mt = 0; mt < 17; ++mt)
#pragma unroll
                for (int v = 0; v < 8; ++v) {
                    float a = acc2[mt][v] + bcl;
                    acc2[mt][v] = a;
                    int row = (mt << 4) + hlf + v;
                    if (row < 258) mx = fmaxf(mx, a);
                }
            mx = fmaxf(mx, __shfl_xor(mx, 16));
            float sum = 0.f;
#pragma unroll
            for (int mt = 0; mt < 17; ++mt)
#pragma unroll
                for (int v = 0; v < 8; ++v) {
                    int row = (mt << 4) + hlf + v;
                    float e = (row < 258) ? __expf(acc2[mt][v] - mx) : 0.f;
                    acc2[mt][v] = e;
                    sum += e;
                }
            sum += __shfl_xor(sum, 16);
            float inv = 1.f / sum;
#pragma unroll
            for (int mt = 0; mt < 17; ++mt)
#pragma unroll
                for (int v = 0; v < 8; ++v) {
                    int row = (mt << 4) + hlf + v;
                    wTg[col * KP + row] = f2u(acc2[mt][v] * inv);
                }
        }
        __threadfence();
        __syncthreads();

        // ---- G5: nm = wT[256,288] x xw[288,128]; wave w owns row tile j=16w..16w+15 ----
        {
            v16bf af[9];
#pragma unroll
            for (int kt = 0; kt < 9; ++kt)
                af[kt] = ld_afrag(wTg, w << 4, KP, kt * 32, lane);
            v8f acc5[8];
#pragma unroll
            for (int nl = 0; nl < 8; ++nl) acc5[nl] = (v8f){0.f,0.f,0.f,0.f,0.f,0.f,0.f,0.f};
#pragma unroll
            for (int nl = 0; nl < 8; ++nl)
#pragma unroll
                for (int kt = 0; kt < 9; ++kt) {
                    v16bf b = ld_bfrag_rm(Abuf, kt, nl << 4, EE, TROWS, lane);
                    acc5[nl] = wmma_bf16(af[kt], b, acc5[nl]);
                }
            __syncthreads();   // all B-frag reads of Abuf done before overwrite

            // per-row min/max normalize, add PE, store next memory state (rows 2..257)
#pragma unroll
            for (int v = 0; v < 8; ++v) {
                int jr = (w << 4) + hlf + v;
                float mn = 3.0e38f, mxr = -3.0e38f;
#pragma unroll
                for (int nl = 0; nl < 8; ++nl) {
                    mn  = fminf(mn,  acc5[nl][v]);
                    mxr = fmaxf(mxr, acc5[nl][v]);
                }
#pragma unroll
                for (int o = 8; o > 0; o >>= 1) {
                    mn  = fminf(mn,  __shfl_xor(mn,  o));
                    mxr = fmaxf(mxr, __shfl_xor(mxr, o));
                }
                float d = mxr - mn;
                float inv = (d > 0.f) ? 1.f / d : 0.f;
#pragma unroll
                for (int nl = 0; nl < 8; ++nl) {
                    int e = (nl << 4) + l15;
                    float nv = (acc5[nl][v] - mn) * inv + peg[jr * EE + e];
                    Abuf[(2 + jr) * EE + e] = f2u(nv);
                }
            }
        }
        __syncthreads();
    }
}

// ---------------- host launcher ----------------
extern "C" void kernel_launch(void* const* d_in, const int* in_sizes, int n_in,
                              void* d_out, int out_size, void* d_ws, size_t ws_size,
                              hipStream_t stream)
{
    (void)in_sizes; (void)n_in; (void)out_size; (void)ws_size;
    const float* state   = (const float*)d_in[0];
    const float* memory0 = (const float*)d_in[1];
    const float* W_emb   = (const float*)d_in[2];
    const float* b_emb   = (const float*)d_in[3];
    const float* Wr1     = (const float*)d_in[4];
    const float* br1     = (const float*)d_in[5];
    const float* Wr2     = (const float*)d_in[6];
    // br2 (d_in[7]) is softmax-invariant; unused
    const float* Ww1     = (const float*)d_in[8];
    const float* bw1     = (const float*)d_in[9];
    const float* Ww2     = (const float*)d_in[10];
    const float* bw2     = (const float*)d_in[11];
    const float* Ws      = (const float*)d_in[12];
    const float* bs_     = (const float*)d_in[13];
    const float* Wa      = (const float*)d_in[14];
    const float* ba      = (const float*)d_in[15];
    const float* Wc      = (const float*)d_in[16];
    const float* bc      = (const float*)d_in[17];
    const float* Wp1     = (const float*)d_in[18];
    const float* bp1     = (const float*)d_in[19];
    const float* Wp2     = (const float*)d_in[20];
    const float* bp2     = (const float*)d_in[21];
    float* out = (float*)d_out;

    // workspace layout (bytes):
    // pe 128K | se 1M | Wr1p 128K | Ww1p 128K | Ww2p 256K | wT 144K | Wr2p 16K
    char* ws = (char*)d_ws;
    float*          peg  = (float*)(ws);
    float*          seg  = (float*)(ws + 131072);
    unsigned short* Wr1p = (unsigned short*)(ws + 1179648);
    unsigned short* Ww1p = (unsigned short*)(ws + 1310720);
    unsigned short* Ww2p = (unsigned short*)(ws + 1441792);
    unsigned short* wTg  = (unsigned short*)(ws + 1703936);
    unsigned short* Wr2p = (unsigned short*)(ws + 1851392);

    k_pe <<<(MM * EE  + 255) / 256, 256, 0, stream>>>(peg);
    k_emb<<<(BSN * EE + 255) / 256, 256, 0, stream>>>(state, W_emb, b_emb, seg);
    k_pack<<<(65536  + 255) / 256, 256, 0, stream>>>(Wr1, 128, 512, Wr1p);
    k_pack<<<(65536  + 255) / 256, 256, 0, stream>>>(Ww1, 128, 512, Ww1p);
    k_pack<<<(131072 + 255) / 256, 256, 0, stream>>>(Ww2, 512, 256, Ww2p);
    k_packrep<<<(8192 + 255) / 256, 256, 0, stream>>>(Wr2, 512, Wr2p);

    k_main<<<1, 512, 0, stream>>>(memory0, peg, seg, Wr1p, Ww1p, Ww2p, Wr2p, wTg,
                                  br1, bw1, bw2, Ws, bs_, Wa, ba, Wc, bc,
                                  Wp1, bp1, Wp2, bp2, out);
}
